// SpikingSRFFN_9234179687027
// MI455X (gfx1250) — matrix operand order repaired and checked by reading, compiled, not run on hardware
//
#include <hip/hip_runtime.h>
#include <hip/hip_bf16.h>
#include <stdint.h>

// ---------------------------------------------------------------------------
// SpikingSRFFN for MI455X (gfx1250, wave32, WMMA + async global->LDS).
//   h = spikes @ W1^T                       (bf16 WMMA GEMM, store bf16)
//   r = sigmoid(spikes @ Wr^T)              (bf16 WMMA GEMM, sigmoid epilogue)
//   scan over T: mixed = d*mixed + (1-d)*h; gated = (mixed>0.5) ? r : 0
//   out = gated @ W2^T                      (bf16 WMMA GEMM, f32 epilogue)
//
// Round-3 changes:
//   * VGPR staging (which spilled to scratch) replaced by CDNA5 async copy:
//     GLOBAL_LOAD_ASYNC_TO_LDS_B128 (ASYNCcnt) into a DOUBLE-BUFFERED LDS
//     tile. Tile k+1 streams into buf^1 while WMMAs consume buf; one
//     s_wait_asynccnt 0 + barrier per iteration. Zero staging VGPRs.
// ---------------------------------------------------------------------------

typedef __attribute__((ext_vector_type(16))) __bf16 v16bf;
typedef __attribute__((ext_vector_type(8)))  float  v8f;

#define D_MODEL 1024
#define D_FF    4096
#define BATCH   4
#define SEQ     2048
#define DECAYF  0.9f
#define THRESHF 0.5f

// GEMM tiling: 128x128 block tile, K-step 64, 256 threads = 8 waves (2x4).
#define BM  128
#define BN  128
#define BK  64
#define LDT 72                      // padded row stride (bf16); 144B, 16B-mult
#define TILE_ELEMS ((BM + BN) * LDT)       // one double-buffer half
#define TILE_BYTES (TILE_ELEMS * 2)

// ---------------------------------------------------------------------------
// 4 contiguous 16B async copies: INST_OFFSET is added to BOTH the global and
// the LDS address (ISA 08 section 4.4), and our per-thread chunks are
// contiguous in both spaces.
__device__ __forceinline__ void async_copy_4x16(uint32_t lds_addr,
                                                const __bf16* gptr) {
    asm volatile(
        "global_load_async_to_lds_b128 %0, %1, off\n\t"
        "global_load_async_to_lds_b128 %0, %1, off offset:16\n\t"
        "global_load_async_to_lds_b128 %0, %1, off offset:32\n\t"
        "global_load_async_to_lds_b128 %0, %1, off offset:48"
        :: "v"(lds_addr), "v"(gptr) : "memory");
}

__device__ __forceinline__ void wait_async_zero() {
    asm volatile("s_wait_asynccnt 0" ::: "memory");
}

// ---------------------------------------------------------------------------
__global__ void f32_to_bf16_kernel(const float* __restrict__ in,
                                   __bf16* __restrict__ out, int n) {
    int i = blockIdx.x * blockDim.x + threadIdx.x;
    if (i < n) out[i] = (__bf16)in[i];
}

// ---------------------------------------------------------------------------
// C[M,N] = A[M,K] @ Bt[N,K]^T   (A, Bt row-major bf16)
// EPI: 0 = store f32, 1 = store bf16, 2 = sigmoid then store bf16
template <int EPI>
__global__ __launch_bounds__(256)
void gemm_bf16_wmma(const __bf16* __restrict__ A,
                    const __bf16* __restrict__ Bt,
                    void* __restrict__ Cout,
                    int M, int N, int K) {
    // double-buffered tiles: [buf][ A(128xLDT) | B(128xLDT) ]
    __shared__ __align__(16) __bf16 lds[2][TILE_ELEMS];

    const int tid   = threadIdx.x;
    const int lane  = tid & 31;
    const int wave  = tid >> 5;
    const int waveM = wave >> 2;   // 0..1 -> 64-row patch
    const int waveN = wave & 3;    // 0..3 -> 32-col patch

    const int mBase = blockIdx.y * BM;
    const int nBase = blockIdx.x * BN;

    v8f acc[4][2];
    #pragma unroll
    for (int i = 0; i < 4; ++i)
        #pragma unroll
        for (int j = 0; j < 2; ++j)
            acc[i][j] = (v8f){0.f, 0.f, 0.f, 0.f, 0.f, 0.f, 0.f, 0.f};

    // Per-thread copy footprint: 4 contiguous 16B chunks per tile (A and B).
    // chunk0 = tid*4; 8 chunks per 64-wide row -> row = tid/2, k0 = 0 or 32.
    const int cRow = tid >> 1;               // 0..127
    const int cK0  = (tid & 1) << 5;         // 0 or 32 (bf16 elems)

    // 32-bit LDS byte addresses (flat->LDS aperture keeps the wave-relative
    // offset in the low 32 bits).
    const uint32_t ldsBase = (uint32_t)(uintptr_t)&lds[0][0];
    const uint32_t ldsOffA = (uint32_t)((cRow * LDT + cK0) * 2);
    const uint32_t ldsOffB = ldsOffA + (uint32_t)(BM * LDT * 2);

    const __bf16* gA = &A [(size_t)(mBase + cRow) * K + cK0];
    const __bf16* gB = &Bt[(size_t)(nBase + cRow) * K + cK0];

    // ---- prologue: stream tile 0 into buffer 0
    async_copy_4x16(ldsBase + ldsOffA, gA);
    async_copy_4x16(ldsBase + ldsOffB, gB);
    wait_async_zero();
    __syncthreads();

    int buf = 0;
    for (int kb = 0; kb < K; kb += BK) {
        // ---- stream next tile into the other buffer (no wait yet)
        if (kb + BK < K) {
            const uint32_t nb = ldsBase + (uint32_t)((buf ^ 1) * TILE_BYTES);
            async_copy_4x16(nb + ldsOffA, gA + kb + BK);
            async_copy_4x16(nb + ldsOffB, gB + kb + BK);
        }

        // ---- compute from current buffer: two WMMA k-steps of 32
        const __bf16* ldsA = &lds[buf][0];
        const __bf16* ldsB = &lds[buf][BM * LDT];
        #pragma unroll
        for (int ks = 0; ks < 2; ++ks) {
            const int ko = ks * 32;
            v16bf af[4], bfr[2];
            // A fragments: 16x32 bf16 per ISA layout.
            // lanes 0-15 : row M=lane,    elems 0..7 = K 0..7,  8..15 = K 16..23
            // lanes 16-31: row M=lane-16, elems 0..7 = K 8..15, 8..15 = K 24..31
            const int m  = lane & 15;
            const int kh = (lane >> 4) << 3;        // 0 or 8
            #pragma unroll
            for (int i = 0; i < 4; ++i) {
                const __bf16* p = &ldsA[(waveM * 64 + i * 16 + m) * LDT + ko + kh];
                uint4* d = (uint4*)&af[i];
                d[0] = *(const uint4*)(p);          // K kh    .. kh+7
                d[1] = *(const uint4*)(p + 16);     // K kh+16 .. kh+23
            }
            // B fragments: 32x16 bf16; col N = lane&15,
            // lanes 0-15 hold K 0..7,16..23, lanes 16-31 hold K 8..15,24..31 (same as A).
            const int n   = lane & 15;
            const int kb2 = (lane >> 4) << 3;       // 0 or 8
            #pragma unroll
            for (int j = 0; j < 2; ++j) {
                const __bf16* p = &ldsB[(waveN * 32 + j * 16 + n) * LDT + ko + kb2];
                uint4* d = (uint4*)&bfr[j];
                d[0] = *(const uint4*)(p);          // K kb2    .. kb2+7
                d[1] = *(const uint4*)(p + 16);     // K kb2+16 .. kb2+23
            }
            #pragma unroll
            for (int i = 0; i < 4; ++i)
                #pragma unroll
                for (int j = 0; j < 2; ++j)
                    acc[i][j] = __builtin_amdgcn_wmma_f32_16x16x32_bf16(
                        false, af[i], false, bfr[j], (short)0, acc[i][j],
                        false, false);
        }

        // ---- next buffer must be fully resident before anyone reads it
        if (kb + BK < K) wait_async_zero();
        __syncthreads();
        buf ^= 1;
    }

    // ---- epilogue. C/D layout: VGPR v -> row v (lanes 0-15) / v+8 (lanes
    // 16-31), col = lane & 15.
    const int rOff = (lane >> 4) << 3;
    const int col0 = lane & 15;
    #pragma unroll
    for (int i = 0; i < 4; ++i) {
        #pragma unroll
        for (int j = 0; j < 2; ++j) {
            const int row0 = mBase + waveM * 64 + i * 16 + rOff;
            const int col  = nBase + waveN * 32 + j * 16 + col0;
            #pragma unroll
            for (int v = 0; v < 8; ++v) {
                const float  x   = acc[i][j][v];
                const size_t off = (size_t)(row0 + v) * (size_t)N + col;
                if (EPI == 0) {
                    ((float*)Cout)[off] = x;
                } else if (EPI == 1) {
                    ((__bf16*)Cout)[off] = (__bf16)x;
                } else {
                    const float s = 1.0f / (1.0f + __expf(-x));
                    ((__bf16*)Cout)[off] = (__bf16)s;
                }
            }
        }
    }
}

// ---------------------------------------------------------------------------
// Sequential-in-T leaky-integrate + threshold gate. One thread per (b,f) lane,
// coalesced along F. Writes gated IN PLACE over h (each thread owns its column).
__global__ __launch_bounds__(256)
void scan_gate_kernel(__bf16* __restrict__ hg, const __bf16* __restrict__ r) {
    const int idx = blockIdx.x * blockDim.x + threadIdx.x;  // b*F + f
    const int b = idx >> 12;            // / D_FF
    const int f = idx & (D_FF - 1);
    size_t base = (size_t)b * SEQ * D_FF + f;

    float mixed = 0.0f;
    const __bf16 zero = (__bf16)0.0f;
    #pragma unroll 4
    for (int t = 0; t < SEQ; ++t) {
        const size_t o  = base + (size_t)t * D_FF;
        const float  hv = (float)hg[o];
        const __bf16 rv = r[o];
        mixed = DECAYF * mixed + (1.0f - DECAYF) * hv;
        hg[o] = (mixed > THRESHF) ? rv : zero;
    }
}

// ---------------------------------------------------------------------------
extern "C" void kernel_launch(void* const* d_in, const int* in_sizes, int n_in,
                              void* d_out, int out_size, void* d_ws, size_t ws_size,
                              hipStream_t stream) {
    const float* spikes = (const float*)d_in[0];  // [B,T,D]
    const float* W1     = (const float*)d_in[1];  // [F,D]
    const float* W2     = (const float*)d_in[2];  // [D,F]
    const float* Wr     = (const float*)d_in[3];  // [F,D]

    const int    M   = BATCH * SEQ;                  // 8192
    const size_t nSp = (size_t)M * D_MODEL;          // 8.4M
    const size_t nW  = (size_t)D_FF * D_MODEL;       // 4.2M (all three weights)
    const size_t nHF = (size_t)M * D_FF;             // 33.6M

    char* ws = (char*)d_ws;
    __bf16* sp_bf = (__bf16*)ws;  ws += nSp * 2;     // 16 MB
    __bf16* w1_bf = (__bf16*)ws;  ws += nW  * 2;     //  8 MB
    __bf16* w2_bf = (__bf16*)ws;  ws += nW  * 2;     //  8 MB
    __bf16* wr_bf = (__bf16*)ws;  ws += nW  * 2;     //  8 MB
    __bf16* h_bf  = (__bf16*)ws;  ws += nHF * 2;     // 64 MB (reused as gated)
    __bf16* r_bf  = (__bf16*)ws;  ws += nHF * 2;     // 64 MB

    // 1) down-convert inputs to bf16
    f32_to_bf16_kernel<<<(unsigned)((nSp + 255) / 256), 256, 0, stream>>>(spikes, sp_bf, (int)nSp);
    f32_to_bf16_kernel<<<(unsigned)((nW  + 255) / 256), 256, 0, stream>>>(W1, w1_bf, (int)nW);
    f32_to_bf16_kernel<<<(unsigned)((nW  + 255) / 256), 256, 0, stream>>>(W2, w2_bf, (int)nW);
    f32_to_bf16_kernel<<<(unsigned)((nW  + 255) / 256), 256, 0, stream>>>(Wr, wr_bf, (int)nW);

    // 2) h = spikes @ W1^T (bf16), r = sigmoid(spikes @ Wr^T) (bf16)
    dim3 g1(D_FF / BN, M / BM);   // 32 x 64
    gemm_bf16_wmma<1><<<g1, 256, 0, stream>>>(sp_bf, w1_bf, h_bf, M, D_FF, D_MODEL);
    gemm_bf16_wmma<2><<<g1, 256, 0, stream>>>(sp_bf, wr_bf, r_bf, M, D_FF, D_MODEL);

    // 3) temporal scan + gate (gated overwrites h_bf in place)
    scan_gate_kernel<<<(BATCH * D_FF) / 256, 256, 0, stream>>>(h_bf, r_bf);

    // 4) out = gated @ W2^T (f32 output)
    dim3 g2(D_MODEL / BN, M / BM);  // 8 x 64
    gemm_bf16_wmma<0><<<g2, 256, 0, stream>>>(h_bf, w2_bf, d_out, M, D_MODEL, D_FF);
}
